// SSD_23871428231937
// MI455X (gfx1250) — matrix operand order repaired and checked by reading, not compiled, not run on hardware
//
#include <hip/hip_runtime.h>
#include <hip/hip_bf16.h>

typedef __attribute__((ext_vector_type(16))) _Float16 v16h;
typedef __attribute__((ext_vector_type(8)))  float    v8f;

// ---------------- problem constants (match reference) ----------------
constexpr int   kB      = 32;        // batch
constexpr int   kA      = 49056;     // anchors
constexpr int   kC      = 8;         // classes
constexpr int   kCF     = 7;         // foreground classes
constexpr int   kK      = 200;       // pre-NMS topk per class
constexpr int   kM      = kCF * kK;  // 1400 candidates per image
constexpr int   kDET    = 200;       // detections per image
constexpr float kNMSTHR = 0.45f;
constexpr float kLOW    = 0.01f;
constexpr long long kTOTAL = (long long)kB * kA;          // 1,569,792 (divisible by 64)
constexpr int   kTILES  = (int)(kTOTAL / 64);             // 24,528
constexpr int   kK1BLK  = kTILES / 8;                     // 3,066 blocks of 8 waves

// =====================================================================
// Kernel 1: fused softmax (WMMA row-sum) + box decode + clip.
// One wave32 per 64-anchor tile. Each lane owns 2 anchors.
// =====================================================================
__device__ __forceinline__ void emit_anchor(long long a, const float e[8], float rinv,
                                            const float* __restrict__ deltas,
                                            const float* __restrict__ dbox,
                                            float* __restrict__ boxes,
                                            float* __restrict__ fg)
{
    int b  = (int)(a / kA);
    int ai = (int)(a - (long long)b * kA);

    // foreground probabilities, laid out [B, 7, A] for coalesced top-k scans
#pragma unroll
    for (int c = 1; c < kC; ++c)
        fg[((long long)b * kCF + (c - 1)) * kA + ai] = e[c] * rinv;

    // decode (BBOX_W = 10,10,5,5) + clip to [0,1]
    float4 d  = ((const float4*)deltas)[a];
    float4 db = ((const float4*)dbox)[ai];
    float w  = db.z - db.x;
    float h  = db.w - db.y;
    float cx = db.x + 0.5f * w;
    float cy = db.y + 0.5f * h;
    float pcx = d.x * 0.1f * w + cx;
    float pcy = d.y * 0.1f * h + cy;
    float pw  = __expf(d.z * 0.2f) * w;
    float ph  = __expf(d.w * 0.2f) * h;
    float4 o;
    o.x = fminf(fmaxf(pcx - 0.5f * pw, 0.0f), 1.0f);
    o.y = fminf(fmaxf(pcy - 0.5f * ph, 0.0f), 1.0f);
    o.z = fminf(fmaxf(pcx + 0.5f * pw, 0.0f), 1.0f);
    o.w = fminf(fmaxf(pcy + 0.5f * ph, 0.0f), 1.0f);
    ((float4*)boxes)[a] = o;
}

__global__ __launch_bounds__(256) void k1_softmax_decode(
    const float* __restrict__ cls,    // [B,A,8]
    const float* __restrict__ deltas, // [B,A,4]
    const float* __restrict__ dbox,   // [A,4]
    float* __restrict__ boxes,        // ws: [B,A,4]
    float* __restrict__ fg)           // ws: [B,7,A]
{
    __shared__ float sums[8][64];                 // per-wave 16x4 D-tile exchange

    const int lane = threadIdx.x & 31;
    const int wv   = threadIdx.x >> 5;
    const int row  = lane & 15;
    const int half = lane >> 4;                   // 0 or 1
    const long long tile = (long long)blockIdx.x * 8 + wv;
    const long long base = tile * 64;
    // lane<16 supplies A-frag K 0..7 (group 0) and K 16..23 (group 2);
    // lane>=16 supplies K 8..15 (group 1) and K 24..31 (group 3).
    const long long a0 = base + (long long)row * 4 + half;      // groups 0/1
    const long long a1 = a0 + 2;                                // groups 2/3

    // ---- load logits, compute stable exp in f32 ----
    float4 l0a = ((const float4*)cls)[a0 * 2 + 0];
    float4 l0b = ((const float4*)cls)[a0 * 2 + 1];
    float4 l1a = ((const float4*)cls)[a1 * 2 + 0];
    float4 l1b = ((const float4*)cls)[a1 * 2 + 1];

    float m0 = fmaxf(fmaxf(fmaxf(l0a.x, l0a.y), fmaxf(l0a.z, l0a.w)),
                     fmaxf(fmaxf(l0b.x, l0b.y), fmaxf(l0b.z, l0b.w)));
    float m1 = fmaxf(fmaxf(fmaxf(l1a.x, l1a.y), fmaxf(l1a.z, l1a.w)),
                     fmaxf(fmaxf(l1b.x, l1b.y), fmaxf(l1b.z, l1b.w)));

    float e0[8], e1[8];
    e0[0] = __expf(l0a.x - m0); e0[1] = __expf(l0a.y - m0);
    e0[2] = __expf(l0a.z - m0); e0[3] = __expf(l0a.w - m0);
    e0[4] = __expf(l0b.x - m0); e0[5] = __expf(l0b.y - m0);
    e0[6] = __expf(l0b.z - m0); e0[7] = __expf(l0b.w - m0);
    e1[0] = __expf(l1a.x - m1); e1[1] = __expf(l1a.y - m1);
    e1[2] = __expf(l1a.z - m1); e1[3] = __expf(l1a.w - m1);
    e1[4] = __expf(l1b.x - m1); e1[5] = __expf(l1b.y - m1);
    e1[6] = __expf(l1b.z - m1); e1[7] = __expf(l1b.w - m1);

    // ---- A fragment: 16x32 f16, row = lane&15 ----
    v16h af;
#pragma unroll
    for (int i = 0; i < 8; ++i) {
        af[i]     = (_Float16)e0[i];
        af[8 + i] = (_Float16)e1[i];
    }
    // ---- B fragment: 32x16 f16 block-diagonal ones, B[k,n] = (k>>3 == n) ----
    v16h bf;
    const int n  = lane & 15;
    const int kb = (lane < 16) ? 0 : 8;
#pragma unroll
    for (int e = 0; e < 16; ++e) {
        int k = kb + e + ((e >= 8) ? 8 : 0);
        bf[e] = (_Float16)(((k >> 3) == n) ? 1.0f : 0.0f);
    }

    v8f acc = {};
    // D[m,n] = sum of anchor-group n of row m  (segmented reduction on the matrix pipe)
    acc = __builtin_amdgcn_wmma_f32_16x16x32_f16(false, af, false, bf,
                                                 (short)0, acc, false, false);

    // ---- scatter the 16x4 sums through LDS (in-order per wave, no barrier) ----
    if (lane < 4) {
#pragma unroll
        for (int i = 0; i < 8; ++i) sums[wv][i * 4 + lane] = acc[i];          // rows 0..7
    }
    if (lane >= 16 && lane < 20) {
#pragma unroll
        for (int i = 0; i < 8; ++i) sums[wv][(8 + i) * 4 + (lane - 16)] = acc[i]; // rows 8..15
    }
    __builtin_amdgcn_wave_barrier();
    asm volatile("" ::: "memory");

    float s0 = sums[wv][row * 4 + half];
    float s1 = sums[wv][row * 4 + half + 2];

    emit_anchor(a0, e0, __builtin_amdgcn_rcpf(s0), deltas, dbox, boxes, fg);
    emit_anchor(a1, e1, __builtin_amdgcn_rcpf(s1), deltas, dbox, boxes, fg);
}

// =====================================================================
// Kernel 2: per (image, fg-class) top-200 via threshold binary search.
// One 256-thread block per (b,c). Data is L2-resident (44 MB < 192 MB).
// =====================================================================
__global__ __launch_bounds__(256) void k2_topk(
    const float* __restrict__ fg,     // [B,7,A]
    const float* __restrict__ boxes,  // [B,A,4]
    float* __restrict__ candS,        // [B,1400]
    float* __restrict__ candB)        // [B,1400,4]
{
    const int bc = blockIdx.x;
    const int b  = bc / kCF;
    const int c  = bc % kCF;
    const float* __restrict__ s = fg + (long long)bc * kA;

    __shared__ int scnt;
    float lo = 0.0f, hi = 1.0f;       // probs are in (0,1); count(s>0) = A >= 200

    for (int it = 0; it < 30; ++it) {
        float mid = 0.5f * (lo + hi);
        if (threadIdx.x == 0) scnt = 0;
        __syncthreads();
        int lc = 0;
        for (int t = threadIdx.x; t < kA; t += 256) {
            __builtin_prefetch(s + t + 256 * 8, 0, 0);
            lc += (s[t] > mid) ? 1 : 0;
        }
        atomicAdd(&scnt, lc);
        __syncthreads();
        int cnt = scnt;
        if (cnt >= kK) lo = mid; else hi = mid;   // invariant: count(>lo) >= 200
        __syncthreads();
    }

    __shared__ int wcnt;
    if (threadIdx.x == 0) wcnt = 0;
    __syncthreads();

    float* myS = candS + (long long)b * kM + c * kK;
    float* myB = candB + ((long long)b * kM + c * kK) * 4;

    for (int t = threadIdx.x; t < kA; t += 256) {
        float v = s[t];
        if (v > lo) {
            int p = atomicAdd(&wcnt, 1);
            if (p < kK) {
                myS[p] = v;
                const float* bx = boxes + ((long long)b * kA + t) * 4;
                myB[p * 4 + 0] = bx[0];
                myB[p * 4 + 1] = bx[1];
                myB[p * 4 + 2] = bx[2];
                myB[p * 4 + 3] = bx[3];
            }
        }
    }
    __syncthreads();
    int filled = wcnt < kK ? wcnt : kK;
    for (int r = threadIdx.x; r < kK; r += 256) {
        if (r >= filled) {
            myS[r] = -__builtin_inff();
            myB[r * 4 + 0] = 0.0f; myB[r * 4 + 1] = 0.0f;
            myB[r * 4 + 2] = 0.0f; myB[r * 4 + 3] = 0.0f;
        }
    }
}

// =====================================================================
// Kernel 3: per-image class-aware NMS, entirely in LDS (~48 KB).
// Candidates staged via ASYNC global->LDS copies; bitonic sort 2048 keys
// descending; greedy suppression; survivors in order are the final top-200.
// Class offsets eliminated: cross-class IoU is provably 0 (offset >= 3 while
// boxes live in [0,1]), so suppression applies only when labels match.
// =====================================================================
constexpr int kNS = 2048;

__global__ __launch_bounds__(512) void k3_nms(
    const float* __restrict__ candS,  // [B,1400]
    const float* __restrict__ candB,  // [B,1400,4]
    float* __restrict__ out)          // boxes[B,200,4] ++ scores[B,200] ++ labels[B,200]
{
    __shared__ float         sl[kM];        // candidate scores   (candidate-id order)
    __shared__ float         bxl[kM][4];    // candidate boxes    (candidate-id order)
    __shared__ float         aarea[kM];     // candidate areas    (candidate-id order)
    __shared__ float         key[kNS];      // sort keys
    __shared__ short         cid[kNS];      // sort payload: candidate id
    __shared__ unsigned char keep[kM];      // sorted-order keep flags
    __shared__ short         sel[kDET];
    __shared__ int           selcnt;

    const int b   = blockIdx.x;
    const int tid = threadIdx.x;
    const float* __restrict__ S  = candS + (long long)b * kM;
    const float* __restrict__ Bx = candB + (long long)b * kM * 4;

    // ---- stage candidates into LDS with async copies (ASYNCcnt path) ----
    for (int i = tid; i < kM; i += 512) {
        unsigned ldsS = (unsigned)(size_t)&sl[i];
        unsigned ldsB = (unsigned)(size_t)&bxl[i][0];
        unsigned offS = (unsigned)i * 4u;
        unsigned offB = (unsigned)i * 16u;
        asm volatile("global_load_async_to_lds_b32 %0, %1, %2"
                     :: "v"(ldsS), "v"(offS), "s"(S) : "memory");
        asm volatile("global_load_async_to_lds_b128 %0, %1, %2"
                     :: "v"(ldsB), "v"(offB), "s"(Bx) : "memory");
    }
    asm volatile("s_wait_asynccnt 0" ::: "memory");
    __syncthreads();

    // keys (low-score -> -inf, matches reference masking), pad to 2048; areas
    for (int i = tid; i < kNS; i += 512) {
        if (i < kM) {
            float v = sl[i];
            key[i] = (v > kLOW) ? v : -__builtin_inff();
        } else {
            key[i] = -__builtin_inff();
        }
        cid[i] = (short)i;
    }
    for (int i = tid; i < kM; i += 512) {
        aarea[i] = (bxl[i][2] - bxl[i][0]) * (bxl[i][3] - bxl[i][1]);
    }
    __syncthreads();

    // bitonic sort, descending
    for (int k = 2; k <= kNS; k <<= 1) {
        for (int j = k >> 1; j > 0; j >>= 1) {
            for (int i = tid; i < kNS; i += 512) {
                int ixj = i ^ j;
                if (ixj > i) {
                    bool up = ((i & k) == 0);           // descending overall
                    float a = key[i], bb = key[ixj];
                    bool sw = up ? (a < bb) : (a > bb);
                    if (sw) {
                        key[i] = bb; key[ixj] = a;
                        short t = cid[i]; cid[i] = cid[ixj]; cid[ixj] = t;
                    }
                }
            }
            __syncthreads();
        }
    }

    for (int i = tid; i < kM; i += 512)
        keep[i] = (key[i] > -1e30f) ? 1 : 0;            // finite score
    __syncthreads();

    // greedy suppression in sorted order (same-class only)
    for (int i = 0; i < kM; ++i) {
        if (keep[i]) {
            int   ci  = cid[i];
            int   li  = ci / kK;                         // class of survivor
            float xi1 = bxl[ci][0], yi1 = bxl[ci][1];
            float xi2 = bxl[ci][2], yi2 = bxl[ci][3];
            float ai  = aarea[ci];
            for (int j = i + 1 + tid; j < kM; j += 512) {
                if (keep[j]) {
                    int cj = cid[j];
                    if (cj / kK == li) {
                        float xl = fmaxf(xi1, bxl[cj][0]);
                        float yt = fmaxf(yi1, bxl[cj][1]);
                        float xr = fminf(xi2, bxl[cj][2]);
                        float yb = fminf(yi2, bxl[cj][3]);
                        float inter = fmaxf(xr - xl, 0.0f) * fmaxf(yb - yt, 0.0f);
                        float iou   = inter / (ai + aarea[cj] - inter);
                        if (iou > kNMSTHR) keep[j] = 0;
                    }
                }
            }
        }
        __syncthreads();
    }

    // survivors in sorted order == final top-200 (array already score-sorted)
    if (tid == 0) {
        int r = 0;
        for (int i = 0; i < kM && r < kDET; ++i)
            if (keep[i]) sel[r++] = (short)i;
        selcnt = r;
    }
    __syncthreads();

    float* oB = out + (long long)b * kDET * 4;
    float* oS = out + (long long)kB * kDET * 4 + (long long)b * kDET;
    float* oL = out + (long long)kB * kDET * 5 + (long long)b * kDET;
    for (int r = tid; r < kDET; r += 512) {
        if (r < selcnt) {
            int i = sel[r];
            int c = cid[i];
            oB[r * 4 + 0] = bxl[c][0];
            oB[r * 4 + 1] = bxl[c][1];
            oB[r * 4 + 2] = bxl[c][2];
            oB[r * 4 + 3] = bxl[c][3];
            oS[r] = sl[c];
            oL[r] = (float)(c / kK + 1);
        } else {
            oB[r * 4 + 0] = 0.0f; oB[r * 4 + 1] = 0.0f;
            oB[r * 4 + 2] = 0.0f; oB[r * 4 + 3] = 0.0f;
            oS[r] = 0.0f;
            oL[r] = 0.0f;
        }
    }
}

// =====================================================================
static inline size_t align_up(size_t x, size_t a) { return (x + a - 1) & ~(a - 1); }

extern "C" void kernel_launch(void* const* d_in, const int* in_sizes, int n_in,
                              void* d_out, int out_size, void* d_ws, size_t ws_size,
                              hipStream_t stream) {
    const float* cls    = (const float*)d_in[0];  // [B,A,8]
    const float* deltas = (const float*)d_in[1];  // [B,A,4]
    const float* dbox   = (const float*)d_in[2];  // [A,4]
    float* out = (float*)d_out;

    char* ws = (char*)d_ws;
    size_t off = 0;
    float* boxes = (float*)(ws + off); off = align_up(off + (size_t)kB * kA * 4 * sizeof(float), 256);
    float* fg    = (float*)(ws + off); off = align_up(off + (size_t)kB * kCF * kA * sizeof(float), 256);
    float* candS = (float*)(ws + off); off = align_up(off + (size_t)kB * kM * sizeof(float), 256);
    float* candB = (float*)(ws + off); off = align_up(off + (size_t)kB * kM * 4 * sizeof(float), 256);
    (void)ws_size; (void)in_sizes; (void)n_in; (void)out_size;

    k1_softmax_decode<<<kK1BLK, 256, 0, stream>>>(cls, deltas, dbox, boxes, fg);
    k2_topk<<<kB * kCF, 256, 0, stream>>>(fg, boxes, candS, candB);
    k3_nms<<<kB, 512, 0, stream>>>(candS, candB, out);
}
